// MultiHeadAttention_40905268527372
// MI455X (gfx1250) — matrix operand order repaired and checked by reading, not compiled, not run on hardware
//
#include <hip/hip_runtime.h>
#include <hip/hip_bf16.h>

// Problem constants (match reference)
constexpr int B_  = 2;
constexpr int L_  = 2048;
constexpr int D_  = 1024;
constexpr int H_  = 16;
constexpr int DK_ = 64;
constexpr int DV_ = 64;

typedef __attribute__((ext_vector_type(16))) __bf16 v16bf;
typedef __attribute__((ext_vector_type(8)))  __bf16 v8bf;
typedef __attribute__((ext_vector_type(8)))  float  v8f;

__device__ __forceinline__ int lane_id() { return (int)(threadIdx.x & 31); }

// ---------------------------------------------------------------------------
// WMMA fragment loaders (layouts per CDNA5 ISA §7.12.2, wave32).
// All loads are contiguous 16/32-byte vectors -> global_load_b128.
// ---------------------------------------------------------------------------

// A fragment 16x32 (MxK) from row-major bf16 source at (row0, k0), ld = lda.
__device__ __forceinline__ v16bf load_A_bf16(const __bf16* __restrict__ A, int lda) {
  const int l = lane_id();
  const int m = l & 15;
  const int half = l >> 4;
  const v8bf lo = *(const v8bf*)(A + (size_t)m * lda + half * 8);
  const v8bf hi = *(const v8bf*)(A + (size_t)m * lda + 16 + half * 8);
  union { v16bf v; v8bf h[2]; } u;
  u.h[0] = lo; u.h[1] = hi;
  return u.v;
}

// Same fragment from a row-major f32 source (convert on load; v_cvt_pk_bf16_f32).
__device__ __forceinline__ v16bf load_A_f32(const float* __restrict__ A, int lda) {
  const int l = lane_id();
  const int m = l & 15;
  const int half = l >> 4;
  const v8f lo = *(const v8f*)(A + (size_t)m * lda + half * 8);
  const v8f hi = *(const v8f*)(A + (size_t)m * lda + 16 + half * 8);
  v16bf a;
#pragma unroll
  for (int i = 0; i < 8; ++i) {
    a[i]     = (__bf16)lo[i];
    a[8 + i] = (__bf16)hi[i];
  }
  return a;
}

// B fragment 32x16 (KxN) from an N-major (transposed) source:
// logical B[k][n] = Bt[n*ldk + k]. Per lane: two 8-element bf16 runs at k = half*8 and 16 + half*8.
__device__ __forceinline__ v16bf load_B_T(const __bf16* __restrict__ Bt, int ldk) {
  const int l = lane_id();
  const int n = l & 15;
  const int half = l >> 4;
  const v8bf lo = *(const v8bf*)(Bt + (size_t)n * ldk + half * 8);
  const v8bf hi = *(const v8bf*)(Bt + (size_t)n * ldk + 16 + half * 8);
  union { v16bf v; v8bf h[2]; } u;
  u.h[0] = lo; u.h[1] = hi;
  return u.v;
}

__device__ __forceinline__ v8f wmma_bf16(v16bf a, v16bf b, v8f c) {
  return __builtin_amdgcn_wmma_f32_16x16x32_bf16(
      /*neg_a=*/false, a, /*neg_b=*/false, b,
      /*c_mod=*/(short)0, c, /*reuse_a=*/false, /*reuse_b=*/false);
}

// C/D mapping: VGPR r, lanes 0-15 -> (M=r, N=lane); lanes 16-31 -> (M=r+8, N=lane-16)
__device__ __forceinline__ void cd_coords(int& row_off, int& col_off) {
  const int l = lane_id();
  row_off = 8 * (l >> 4);
  col_off = l & 15;
}

// ---------------------------------------------------------------------------
// Kernels
// ---------------------------------------------------------------------------

// Weights: f32 [K x N] -> bf16 transposed [N x K]
__global__ void cvt_transpose_kernel(const float* __restrict__ in, __bf16* __restrict__ out,
                                     int K, int N) {
  int idx = blockIdx.x * 256 + threadIdx.x;
  if (idx < K * N) {
    int k = idx / N, n = idx % N;
    out[(size_t)n * K + k] = (__bf16)in[idx];
  }
}

// Projection: Y = X[M=4096 x K=1024] * W (W given N-major bf16), scaled, written in
// per-head layout. TRANSPOSED=0: Yh[B,H,L,64]; TRANSPOSED=1: Yh[B,H,64,L].
// block=128 (4 waves); per wave: 32 M-rows x 64 N -> 8 WMMAs/k-step, B reused x2.
template <int TRANSPOSED>
__global__ void proj_kernel(const float* __restrict__ X, const __bf16* __restrict__ Wt,
                            __bf16* __restrict__ Yh, float scale) {
  constexpr int K = D_;
  const int wave  = threadIdx.x >> 5;
  const int mbase = blockIdx.x * 128 + wave * 32;
  const int nbase = blockIdx.y * 64;
  v8f acc[2][4] = {};
  for (int kb = 0; kb < K; kb += 32) {
    __builtin_prefetch(X + (size_t)mbase * K + kb + 64, 0, 1);
    v16bf a0 = load_A_f32(X + (size_t)mbase * K + kb, K);
    v16bf a1 = load_A_f32(X + (size_t)(mbase + 16) * K + kb, K);
#pragma unroll
    for (int nt = 0; nt < 4; ++nt) {
      v16bf bf = load_B_T(Wt + (size_t)(nbase + nt * 16) * K + kb, K);
      acc[0][nt] = wmma_bf16(a0, bf, acc[0][nt]);
      acc[1][nt] = wmma_bf16(a1, bf, acc[1][nt]);
    }
  }
  int row_off, col_off;
  cd_coords(row_off, col_off);
  // Blocks never straddle a batch boundary (2048 % 128 == 0): bb/ll0 uniform.
  const int bb  = mbase / L_;
  const int ll0 = (mbase % L_) + row_off;
#pragma unroll
  for (int nt = 0; nt < 4; ++nt) {
    // Each 16-wide N-tile lies inside one head (DK=64, 16-aligned).
    const int n0  = nbase + nt * 16;
    const int hh  = n0 / DK_;
    const int dk  = (n0 % DK_) + col_off;
    __bf16* basePtr = TRANSPOSED
        ? Yh + (((size_t)bb * H_ + hh) * DK_ + dk) * L_        // [B,H,DK,L], + ll
        : Yh + (((size_t)bb * H_ + hh) * L_) * DK_ + dk;       // [B,H,L,DK], + ll*DK
#pragma unroll
    for (int mi = 0; mi < 2; ++mi) {
#pragma unroll
      for (int r = 0; r < 8; ++r) {
        const int ll = ll0 + mi * 16 + r;
        const __bf16 val = (__bf16)(acc[mi][nt][r] * scale);
        if (TRANSPOSED) basePtr[ll] = val;
        else            basePtr[(size_t)ll * DK_] = val;
      }
    }
  }
}

// scores = Qh * Kh^T (per b,h), mask applied, f32 into attn region.
// Kh row-major [L x DK] is exactly the N-major layout for the B operand.
__global__ void scores_kernel(const __bf16* __restrict__ Qh, const __bf16* __restrict__ Kh,
                              const unsigned char* __restrict__ mask, float* __restrict__ attn) {
  const int bh = blockIdx.z;
  const int b  = bh / H_;
  const __bf16* Q  = Qh + (size_t)bh * L_ * DK_;
  const __bf16* Kp = Kh + (size_t)bh * L_ * DK_;
  float* S = attn + (size_t)bh * L_ * L_;
  const unsigned char* Mk = mask + (size_t)b * L_ * L_;

  const int wave  = threadIdx.x >> 5;
  const int qbase = blockIdx.x * 128 + wave * 32;
  const int kbase = blockIdx.y * 64;
  v8f acc[2][4] = {};
#pragma unroll
  for (int kk = 0; kk < DK_; kk += 32) {
    v16bf a0 = load_A_bf16(Q + (size_t)qbase * DK_ + kk, DK_);
    v16bf a1 = load_A_bf16(Q + (size_t)(qbase + 16) * DK_ + kk, DK_);
#pragma unroll
    for (int nt = 0; nt < 4; ++nt) {
      v16bf bf = load_B_T(Kp + (size_t)(kbase + nt * 16) * DK_ + kk, DK_);
      acc[0][nt] = wmma_bf16(a0, bf, acc[0][nt]);
      acc[1][nt] = wmma_bf16(a1, bf, acc[1][nt]);
    }
  }
  int row_off, col_off;
  cd_coords(row_off, col_off);
#pragma unroll
  for (int mi = 0; mi < 2; ++mi) {
#pragma unroll
    for (int nt = 0; nt < 4; ++nt) {
      const int kc = kbase + nt * 16 + col_off;
#pragma unroll
      for (int r = 0; r < 8; ++r) {
        const int q = qbase + mi * 16 + r + row_off;
        const float v = acc[mi][nt][r];
        S[(size_t)q * L_ + kc] = Mk[(size_t)q * L_ + kc] ? v : -1e9f;
      }
    }
  }
}

// Row softmax over last axis, in place. One block (256 thr) per row of 2048.
__global__ void softmax_kernel(float* __restrict__ attn) {
  float* p = attn + (size_t)blockIdx.x * L_;
  __shared__ float red[256];
  const int t = threadIdx.x;
  float lmax = -3.4e38f;
  for (int i = t; i < L_; i += 256) lmax = fmaxf(lmax, p[i]);
  red[t] = lmax; __syncthreads();
  for (int s = 128; s > 0; s >>= 1) {
    if (t < s) red[t] = fmaxf(red[t], red[t + s]);
    __syncthreads();
  }
  const float rmax = red[0]; __syncthreads();
  float lsum = 0.f;
  for (int i = t; i < L_; i += 256) {
    float e = __expf(p[i] - rmax);
    p[i] = e;
    lsum += e;
  }
  red[t] = lsum; __syncthreads();
  for (int s = 128; s > 0; s >>= 1) {
    if (t < s) red[t] += red[t + s];
    __syncthreads();
  }
  const float inv = 1.0f / red[0];
  for (int i = t; i < L_; i += 256) p[i] *= inv;
}

// ctx_h = attn(f32->bf16 on load) * Vh ; VhT is [B,H,DV,L] (N-major for B operand).
// Result into ctx[B, L, H*DV] bf16 (row-major -> contiguous A loads for oproj).
__global__ void ctx_kernel(const float* __restrict__ attn, const __bf16* __restrict__ VhT,
                           __bf16* __restrict__ ctx) {
  const int bh = blockIdx.z;
  const int b = bh / H_, h = bh % H_;
  const float*  A = attn + (size_t)bh * L_ * L_;
  const __bf16* V = VhT  + (size_t)bh * DV_ * L_;   // [DV][L]

  const int wave  = threadIdx.x >> 5;
  const int qbase = blockIdx.x * 128 + wave * 32;
  v8f acc[2][4] = {};
  for (int kk = 0; kk < L_; kk += 32) {
    __builtin_prefetch(A + (size_t)qbase * L_ + kk + 64, 0, 1);
    v16bf a0 = load_A_f32(A + (size_t)qbase * L_ + kk, L_);
    v16bf a1 = load_A_f32(A + (size_t)(qbase + 16) * L_ + kk, L_);
#pragma unroll
    for (int nt = 0; nt < 4; ++nt) {
      v16bf bf = load_B_T(V + (size_t)(nt * 16) * L_ + kk, L_);
      acc[0][nt] = wmma_bf16(a0, bf, acc[0][nt]);
      acc[1][nt] = wmma_bf16(a1, bf, acc[1][nt]);
    }
  }
  int row_off, col_off;
  cd_coords(row_off, col_off);
#pragma unroll
  for (int mi = 0; mi < 2; ++mi) {
#pragma unroll
    for (int nt = 0; nt < 4; ++nt) {
      const int dv = nt * 16 + col_off;
#pragma unroll
      for (int r = 0; r < 8; ++r) {
        const int qq = qbase + mi * 16 + r + row_off;
        ctx[((size_t)b * L_ + qq) * (H_ * DV_) + h * DV_ + dv] = (__bf16)acc[mi][nt][r];
      }
    }
  }
}

// out = ctx[4096x1024]*Wo + residual (Wo given N-major bf16; f32 into d_out out region)
__global__ void oproj_kernel(const __bf16* __restrict__ ctx, const __bf16* __restrict__ WoT,
                             const float* __restrict__ residual, float* __restrict__ out) {
  constexpr int K = H_ * DV_;
  constexpr int N = D_;
  const int wave  = threadIdx.x >> 5;
  const int mbase = blockIdx.x * 128 + wave * 32;
  const int nbase = blockIdx.y * 64;
  v8f acc[2][4] = {};
  for (int kb = 0; kb < K; kb += 32) {
    __builtin_prefetch(ctx + (size_t)mbase * K + kb + 32, 0, 1);
    v16bf a0 = load_A_bf16(ctx + (size_t)mbase * K + kb, K);
    v16bf a1 = load_A_bf16(ctx + (size_t)(mbase + 16) * K + kb, K);
#pragma unroll
    for (int nt = 0; nt < 4; ++nt) {
      v16bf bf = load_B_T(WoT + (size_t)(nbase + nt * 16) * K + kb, K);
      acc[0][nt] = wmma_bf16(a0, bf, acc[0][nt]);
      acc[1][nt] = wmma_bf16(a1, bf, acc[1][nt]);
    }
  }
  int row_off, col_off;
  cd_coords(row_off, col_off);
#pragma unroll
  for (int mi = 0; mi < 2; ++mi) {
#pragma unroll
    for (int nt = 0; nt < 4; ++nt) {
      const int n = nbase + nt * 16 + col_off;
#pragma unroll
      for (int r = 0; r < 8; ++r) {
        const int m = mbase + mi * 16 + r + row_off;
        out[(size_t)m * N + n] = acc[mi][nt][r] + residual[(size_t)m * N + n];
      }
    }
  }
}

// LayerNorm over D=1024 per row, in place on d_out out region.
__global__ void layernorm_kernel(float* __restrict__ out, const float* __restrict__ g,
                                 const float* __restrict__ be) {
  float* p = out + (size_t)blockIdx.x * D_;
  __shared__ float s1[256], s2[256];
  const int t = threadIdx.x;
  float a = 0.f, q = 0.f;
  for (int i = t; i < D_; i += 256) {
    float x = p[i];
    a += x; q += x * x;
  }
  s1[t] = a; s2[t] = q; __syncthreads();
  for (int s = 128; s > 0; s >>= 1) {
    if (t < s) { s1[t] += s1[t + s]; s2[t] += s2[t + s]; }
    __syncthreads();
  }
  const float mu  = s1[0] * (1.0f / D_);
  const float var = s2[0] * (1.0f / D_) - mu * mu;
  const float rst = rsqrtf(var + 1e-6f);
  for (int i = t; i < D_; i += 256)
    p[i] = (p[i] - mu) * rst * g[i] + be[i];
}

// ---------------------------------------------------------------------------
// Launch
// ---------------------------------------------------------------------------
extern "C" void kernel_launch(void* const* d_in, const int* in_sizes, int n_in,
                              void* d_out, int out_size, void* d_ws, size_t ws_size,
                              hipStream_t stream) {
  (void)in_sizes; (void)n_in; (void)out_size; (void)ws_size;

  const float* q    = (const float*)d_in[0];
  const float* k    = (const float*)d_in[1];
  const float* v    = (const float*)d_in[2];
  const unsigned char* mask = (const unsigned char*)d_in[3];
  const float* w_q  = (const float*)d_in[4];
  const float* w_k  = (const float*)d_in[5];
  const float* w_v  = (const float*)d_in[6];
  const float* w_o  = (const float*)d_in[7];
  const float* ln_g = (const float*)d_in[8];
  const float* ln_b = (const float*)d_in[9];

  float* out  = (float*)d_out;                          // [B, L, D]
  float* attn = out + (size_t)B_ * L_ * D_;             // [B, H, L, L]

  // Workspace layout
  char* ws = (char*)d_ws;
  const size_t headElems = (size_t)B_ * H_ * L_ * DK_;  // 4 Mi
  const size_t wElems    = (size_t)D_ * H_ * DK_;       // 1 Mi
  __bf16* Qh  = (__bf16*)(ws);                          // [B,H,L,DK]
  __bf16* Kh  = Qh + headElems;                         // [B,H,L,DK]  (== N-major K^T)
  __bf16* VhT = Kh + headElems;                         // [B,H,DV,L]  (N-major V)
  __bf16* Ctx = VhT + headElems;                        // [B,L,H*DV]
  __bf16* WqT = Ctx + headElems;                        // [N x K]
  __bf16* WkT = WqT + wElems;
  __bf16* WvT = WkT + wElems;
  __bf16* WoT = WvT + wElems;

  // 1) weights -> bf16, transposed to N-major
  {
    const int n = (int)wElems;
    dim3 g((n + 255) / 256), b(256);
    cvt_transpose_kernel<<<g, b, 0, stream>>>(w_q, WqT, D_, H_ * DK_);
    cvt_transpose_kernel<<<g, b, 0, stream>>>(w_k, WkT, D_, H_ * DK_);
    cvt_transpose_kernel<<<g, b, 0, stream>>>(w_v, WvT, D_, H_ * DK_);
    cvt_transpose_kernel<<<g, b, 0, stream>>>(w_o, WoT, H_ * DV_, D_);
  }

  // 2) projections (M=4096, N=1024); Q folded with 1/sqrt(DK); V written transposed
  {
    dim3 g(32, 16), b(128);
    proj_kernel<0><<<g, b, 0, stream>>>(q, WqT, Qh, 0.125f);
    proj_kernel<0><<<g, b, 0, stream>>>(k, WkT, Kh, 1.0f);
    proj_kernel<1><<<g, b, 0, stream>>>(v, WvT, VhT, 1.0f);
  }

  // 3) masked scores into attn region
  {
    dim3 g(L_ / 128, L_ / 64, B_ * H_), b(128);
    scores_kernel<<<g, b, 0, stream>>>(Qh, Kh, mask, attn);
  }

  // 4) softmax, one block per row
  {
    dim3 g(B_ * H_ * L_), b(256);
    softmax_kernel<<<g, b, 0, stream>>>(attn);
  }

  // 5) attn * V -> ctx (bf16)
  {
    dim3 g(L_ / 128, 1, B_ * H_), b(128);
    ctx_kernel<<<g, b, 0, stream>>>(attn, VhT, Ctx);
  }

  // 6) output projection + residual
  {
    dim3 g(32, 16), b(128);
    oproj_kernel<<<g, b, 0, stream>>>(Ctx, WoT, q, out);
  }

  // 7) layernorm in place
  {
    dim3 g(B_ * L_), b(256);
    layernorm_kernel<<<g, b, 0, stream>>>(out, ln_g, ln_b);
  }
}
